// HybridIDMModel_80238579024139
// MI455X (gfx1250) — matrix operand order repaired and checked by reading, not compiled, not run on hardware
//
#include <hip/hip_runtime.h>
#include <hip/hip_bf16.h>

// ---------------------------------------------------------------------------
// Liquid-time-constant RNN scan for MI455X (gfx1250, wave32, WMMA).
//
// Block = 16 batch rows x 8 waves; wave w owns H-columns [16w, 16w+16).
// W_hh^T panel + (W_xh|b_hh) panel live in VGPRs for the whole 256-step scan.
// h (16x128 f16) is double-buffered in LDS, 1 barrier/step.
// Per step per wave: 5 x v_wmma_f32_16x16x32_f16 (input-proj+bias folded via
// augmented-A, 4 chunks for the K=128 recurrent reduction), 8x hardware tanh
// (or branchless exp2/rcp fallback), 8 FMA state update, 8 ds_store_b16,
// 8 ds_load_b128.
// ---------------------------------------------------------------------------

typedef _Float16 v16h __attribute__((ext_vector_type(16)));
typedef _Float16 v8h  __attribute__((ext_vector_type(8)));
typedef float    v8f  __attribute__((ext_vector_type(8)));

#define S_LEN   256
#define H_DIM   128
#define B_SZ    2048
#define DIN     3
#define DOUT    6
#define DT_C    0.1f
#define ROWPAD  136   // halves per LDS row (128 + 8 pad -> 272B, 16B aligned)

__device__ __forceinline__ float softplus_f(float v) {
    return fmaxf(v, 0.0f) + log1pf(__expf(-fabsf(v)));
}

// Branch-free tanh on the scan critical path.
__device__ __forceinline__ float fast_tanh(float v) {
#if __has_builtin(__builtin_amdgcn_tanhf)
    return __builtin_amdgcn_tanhf(v);            // v_tanh_f32
#elif __has_builtin(__builtin_amdgcn_exp2f) && __has_builtin(__builtin_amdgcn_rcpf)
    // tanh(x) = 1 - 2/(exp(2x)+1); saturates to +/-1 at +/-inf, branchless.
    const float t = __builtin_amdgcn_exp2f(v * 2.8853900817779268f); // 2*log2(e)
    return 1.0f - 2.0f * __builtin_amdgcn_rcpf(t + 1.0f);
#else
    const float t = __expf(2.0f * v);
    return 1.0f - 2.0f / (t + 1.0f);
#endif
}

__global__ __launch_bounds__(256) void lnn_scan_kernel(
    const float* __restrict__ x,        // [B, S, 3]
    const float* __restrict__ W_xh,     // [128, 3]
    const float* __restrict__ W_hh,     // [128, 128]
    const float* __restrict__ b_hh,     // [128]
    const float* __restrict__ log_tau,  // [128]
    float* __restrict__ hT)             // [B, 128] scratch (d_ws)
{
    __shared__ __align__(16) _Float16 hbuf[2][16 * ROWPAD];

    const int lane    = threadIdx.x & 31;
    const int wave    = threadIdx.x >> 5;      // 0..7 = N-tile of H
    const int half    = (lane >> 4) & 1;       // lane group 0-15 / 16-31
    const int l16     = lane & 15;
    const int rowbase = blockIdx.x * 16;       // batch rows of this block
    const bool lo_half = (half == 0);

    const int n = wave * 16 + l16;             // H column owned by this lane

    // dtau = DT / (softplus(log_tau[n]) + 1e-3)   (time-invariant)
    const float dtau = DT_C / (softplus_f(log_tau[n]) + 0.001f);

    // B operand for input projection: rows k=0..2 are W_xh^T, k=3 is b_hh.
    // f16 B layout (K=32,N=16): lanes 0-15 hold K 0-7 & 16-23 of column n,
    // lanes 16-31 hold K 8-15 & 24-31 (all zero here).
    v16h bx = {};
    {
        const _Float16 w0 = (_Float16)W_xh[n * DIN + 0];
        const _Float16 w1 = (_Float16)W_xh[n * DIN + 1];
        const _Float16 w2 = (_Float16)W_xh[n * DIN + 2];
        const _Float16 bb = (_Float16)b_hh[n];
        bx[0] = lo_half ? w0 : (_Float16)0.0f;
        bx[1] = lo_half ? w1 : (_Float16)0.0f;
        bx[2] = lo_half ? w2 : (_Float16)0.0f;
        bx[3] = lo_half ? bb : (_Float16)0.0f;
    }

    // Resident B operands for h @ W_hh^T, 4 chunks of K=32.
    // B^T[k][n] = W_hh[n][k]: lane half h takes K 8h..8h+7 and 16+8h..23+8h of row n.
    v16h bw[4];
    #pragma unroll
    for (int kc = 0; kc < 4; ++kc) {
        const float* src = W_hh + (size_t)n * H_DIM + kc * 32 + half * 8;
        #pragma unroll
        for (int i = 0; i < 16; ++i) bw[kc][i] = (_Float16)src[i < 8 ? i : i + 8];
    }

    // Hidden state: 8 f32 per lane; element j sits at (m = j + 8*half, col n)
    v8f h = {};

    // Zero-init buffer 0 (all waves cover disjoint columns).
    #pragma unroll
    for (int j = 0; j < 8; ++j) {
        const int m = j + half * 8;
        hbuf[0][m * ROWPAD + n] = (_Float16)0.0f;
    }

    // x row feeding this lane's A row. Same address for lane L and L+16 so all
    // lanes load (same cacheline); upper half is masked out branchlessly.
    const float* xrow = x + (size_t)(rowbase + l16) * (S_LEN * DIN);
    const float zf = 0.0f;

    __syncthreads();

    for (int s = 0; s < S_LEN; ++s) {
        const _Float16* __restrict__ hb  = hbuf[s & 1];
        _Float16*       __restrict__ hbn = hbuf[(s + 1) & 1];

        // Augmented A for input projection: [x0 x1 x2 1 0...0] (lanes 0-15)
        const float* xs = xrow + s * DIN;
        const float x0 = xs[0], x1 = xs[1], x2 = xs[2];
        if (s + 1 < S_LEN) __builtin_prefetch(xs + DIN, 0, 3);
        v16h ax = {};
        ax[0] = (_Float16)(lo_half ? x0 : zf);
        ax[1] = (_Float16)(lo_half ? x1 : zf);
        ax[2] = (_Float16)(lo_half ? x2 : zf);
        ax[3] = lo_half ? (_Float16)1.0f : (_Float16)0.0f;

        // Preload all 4 A chunks (8 ds_load_b128 issued back-to-back).
        // f16 A layout (M=16,K=32): lane<16 row l16 holds K 0-7 & 16-23,
        // lane>=16 holds K 8-15 & 24-31 -> two 16B LDS loads per chunk.
        v16h a[4];
        #pragma unroll
        for (int kc = 0; kc < 4; ++kc) {
            const _Float16* base = hb + l16 * ROWPAD + kc * 32 + half * 8;
            const v8h lov = *(const v8h*)(base);
            const v8h hiv = *(const v8h*)(base + 16);
            #pragma unroll
            for (int i = 0; i < 8; ++i) { a[kc][i] = lov[i]; a[kc][i + 8] = hiv[i]; }
        }

        // acc = x_t @ W_xh^T + b_hh  (overlaps with the LDS loads above)
        v8f acc = {};
        acc = __builtin_amdgcn_wmma_f32_16x16x32_f16(false, ax, false, bx,
                                                     (short)0, acc, false, false);

        // acc += h_s @ W_hh^T   (K = 128 as 4 chunks of 32)
        #pragma unroll
        for (int kc = 0; kc < 4; ++kc) {
            acc = __builtin_amdgcn_wmma_f32_16x16x32_f16(false, a[kc], false, bw[kc],
                                                         (short)0, acc, false, false);
        }

        // f = tanh(acc); h += (f - h) * DT/tau; publish f16 h_{s+1}
        #pragma unroll
        for (int j = 0; j < 8; ++j) {
            const float f = fast_tanh(acc[j]);
            h[j] += (f - h[j]) * dtau;
            const int m = j + half * 8;
            hbn[m * ROWPAD + n] = (_Float16)h[j];
        }

        __syncthreads();
    }

    // Spill final f32 hidden state for the head kernel.
    #pragma unroll
    for (int j = 0; j < 8; ++j) {
        const int m = j + half * 8;
        hT[(size_t)(rowbase + m) * H_DIM + n] = h[j];
    }
}

__global__ __launch_bounds__(256) void lnn_head_kernel(
    const float* __restrict__ hT,    // [B, 128]
    const float* __restrict__ fc_W,  // [6, 128]
    const float* __restrict__ fc_b,  // [6]
    float* __restrict__ out)         // [B, 6]
{
    const int b = blockIdx.x * blockDim.x + threadIdx.x;
    if (b >= B_SZ) return;

    const float* hr = hT + (size_t)b * H_DIM;
    float acc[DOUT];
    #pragma unroll
    for (int o = 0; o < DOUT; ++o) acc[o] = fc_b[o];

    for (int k = 0; k < H_DIM; ++k) {
        const float hv = hr[k];
        #pragma unroll
        for (int o = 0; o < DOUT; ++o) acc[o] = fmaf(hv, fc_W[o * H_DIM + k], acc[o]);
    }

    #pragma unroll
    for (int o = 0; o < DOUT; ++o) out[(size_t)b * DOUT + o] = softplus_f(acc[o]);
}

extern "C" void kernel_launch(void* const* d_in, const int* in_sizes, int n_in,
                              void* d_out, int out_size, void* d_ws, size_t ws_size,
                              hipStream_t stream) {
    const float* x       = (const float*)d_in[0];  // [2048, 256, 3]
    const float* W_xh    = (const float*)d_in[1];  // [128, 3]
    const float* W_hh    = (const float*)d_in[2];  // [128, 128]
    const float* b_hh    = (const float*)d_in[3];  // [128]
    const float* log_tau = (const float*)d_in[4];  // [128]
    const float* fc_W    = (const float*)d_in[5];  // [6, 128]
    const float* fc_b    = (const float*)d_in[6];  // [6]
    float* out = (float*)d_out;                    // [2048, 6]
    float* hT  = (float*)d_ws;                     // [2048, 128] = 1 MB scratch

    lnn_scan_kernel<<<B_SZ / 16, 256, 0, stream>>>(x, W_xh, W_hh, b_hh, log_tau, hT);
    lnn_head_kernel<<<(B_SZ + 255) / 256, 256, 0, stream>>>(hT, fc_W, fc_b, out);
}
